// Pos_att_56049323213171
// MI455X (gfx1250) — matrix operand order repaired and checked by reading, not compiled, not run on hardware
//
#include <hip/hip_runtime.h>

// Pos_att for MI455X (gfx1250): bf16 WMMA flash-style attention.
// y = conv1x1(x) in bf16 [N=9216][32]; S = y yT (never materialized);
// rowmax/rowscale precomputed; x6 = x5 @ softmax(S) via WMMA with LDS-staged
// P tiles and TDM-staged A tiles; final softmax over W.

#define HH 96
#define WWID 96
#define NN 9216
#define CC 256
#define C8 32

typedef __attribute__((ext_vector_type(16))) __bf16       bf16x16;
typedef __attribute__((ext_vector_type(8)))  float        f32x8;
typedef __attribute__((ext_vector_type(4)))  unsigned int u32x4;
typedef __attribute__((ext_vector_type(4)))  float        f32x4;
typedef __attribute__((ext_vector_type(8)))  int          i32x8;
typedef __attribute__((ext_vector_type(4)))  int          i32x4;

union BV { u32x4 q[2]; bf16x16 v; };

__device__ __forceinline__ unsigned int pack_bf16(float lo, float hi) {
  unsigned short l = __builtin_bit_cast(unsigned short, (__bf16)lo);
  unsigned short h = __builtin_bit_cast(unsigned short, (__bf16)hi);
  return (unsigned int)l | ((unsigned int)h << 16);
}

// TDM: async DMA of a 2D tile (tile_dim0=32 bf16 = 64B per row, 256 rows,
// row stride 9216 elements) from global xb into LDS. D# per
// CDNA5 ISA (group0: count/lds_addr/global_addr/type=2;
// group1: data_size=2B, dims, stride; groups 2/3 zero for a 2D tile).
__device__ __forceinline__ void tdm_load_A(const __bf16* xb, int i0,
                                           unsigned int ldsOff) {
  unsigned long long ga =
      (unsigned long long)(uintptr_t)xb + (unsigned long long)i0 * 2ull;
  u32x4 g0 = { 1u,                                  // count=1 (valid)
               ldsOff,                              // lds_addr
               (unsigned int)ga,                    // global_addr[31:0]
               (((unsigned int)(ga >> 32)) & 0x01FFFFFFu) | 0x80000000u };
  i32x8 g1 = { 0x00010000,                          // data_size=1 (2 bytes)
               (int)(32u << 16),                    // tensor_dim0 = 32
               (int)(256u << 16),                   // tensor_dim1 = 256
               (int)(32u << 16),                    // tile_dim0 = 32
               256,                                 // tile_dim1 = 256
               9216,                                // tensor_dim0_stride lo32
               0, 0 };
  i32x4 gz = { 0, 0, 0, 0 };
#if defined(__clang_major__) && __clang_major__ >= 23
  i32x8 gz8 = { 0, 0, 0, 0, 0, 0, 0, 0 };
  __builtin_amdgcn_tensor_load_to_lds(g0, g1, gz, gz, gz8, 0);
#else
  __builtin_amdgcn_tensor_load_to_lds(g0, g1, gz, gz, 0);
#endif
}

// ---------------------------------------------------------------------------
// Kernel 1: 1x1 conv (32x256 @ 256x9216) -> y_bf [p=w*H+h][c8] bf16,
//           plus bf16 copy of x (xb = x5 layout [C][q=h*W+w]).
// ---------------------------------------------------------------------------
__global__ __launch_bounds__(256) void k_conv(const float* __restrict__ x,
                                              const float* __restrict__ w,
                                              const float* __restrict__ b,
                                              __bf16* __restrict__ ybf,
                                              __bf16* __restrict__ xb) {
  __shared__ float  wl[C8 * 257];      // padded to dodge bank conflicts
  __shared__ __bf16 xs[CC * 32];
  const int t  = threadIdx.x;
  const int q0 = blockIdx.x * 32;

  for (int rep = 0; rep < C8; ++rep)
    wl[rep * 257 + t] = w[rep * 256 + t];

  for (int rep = 0; rep < 32; ++rep) {
    int c  = rep * 8 + (t >> 5);
    int qq = t & 31;
    float v = x[(size_t)c * NN + q0 + qq];
    __bf16 bv = (__bf16)v;
    xs[c * 32 + qq] = bv;
    xb[(size_t)c * NN + q0 + qq] = bv;
  }
  __syncthreads();

  const int c8 = t & 31;
  const int qq = t >> 5;               // 0..7
  float a0 = 0.f, a1 = 0.f, a2 = 0.f, a3 = 0.f;
  for (int c = 0; c < CC; ++c) {
    float wv = wl[c8 * 257 + c];
    a0 = fmaf(wv, (float)xs[c * 32 + qq     ], a0);
    a1 = fmaf(wv, (float)xs[c * 32 + qq +  8], a1);
    a2 = fmaf(wv, (float)xs[c * 32 + qq + 16], a2);
    a3 = fmaf(wv, (float)xs[c * 32 + qq + 24], a3);
  }
  float bias = b[c8];
  float accs[4] = {a0, a1, a2, a3};
#pragma unroll
  for (int p = 0; p < 4; ++p) {
    int q = q0 + qq + 8 * p;
    int h = q / WWID, wcol = q % WWID;
    int pi = wcol * HH + h;            // p-ordering (w*H + h)
    ybf[(size_t)pi * C8 + c8] = (__bf16)(accs[p] + bias);
  }
}

// ---------------------------------------------------------------------------
// Kernel 2: row stats of S = y yT. One block per 16 rows; 4 waves split the
// 576 column tiles; one V_WMMA_F32_16X16X32_BF16 per 16x16 S tile (K=32=C8).
// ---------------------------------------------------------------------------
__global__ __launch_bounds__(128) void k_rowstats(const __bf16* __restrict__ ybf,
                                                  float* __restrict__ rowmax,
                                                  float* __restrict__ rowscale) {
  __shared__ float pm[4][16];
  __shared__ float ps[4][16];
  const int lane = threadIdx.x & 31;
  const int ww   = threadIdx.x >> 5;
  const int i0   = blockIdx.x * 16;
  const int mrow = lane & 15;
  const int half = lane >> 4;

  // A tile: rows i0..i0+15, K=0..31 (ISA 16-bit A layout: two 16B chunks/lane)
  BV A;
  {
    const char* ab = (const char*)ybf + (size_t)(i0 + mrow) * 64 + half * 16;
    A.q[0] = *(const u32x4*)(ab);
    A.q[1] = *(const u32x4*)(ab + 32);
  }

  float mx[8], sm[8];
#pragma unroll
  for (int r = 0; r < 8; ++r) { mx[r] = -__builtin_inff(); sm[r] = 0.f; }

  for (int jt = 0; jt < 144; ++jt) {
    int j0 = (ww * 144 + jt) * 16;
    BV Bm;   // B[k][n] = y[j0+n][k]; lane=col, K in two 8-chunks per half (same as A)
    const char* bb = (const char*)ybf + (size_t)(j0 + mrow) * 64 + half * 16;
    Bm.q[0] = *(const u32x4*)(bb);
    Bm.q[1] = *(const u32x4*)(bb + 32);
    f32x8 z = {};
    f32x8 s = __builtin_amdgcn_wmma_f32_16x16x32_bf16(false, A.v, false, Bm.v,
                                                      (short)0, z, false, false);
#pragma unroll
    for (int r = 0; r < 8; ++r) {
      float v  = s[r];
      float nm = fmaxf(mx[r], v);
      sm[r] = sm[r] * __expf(mx[r] - nm) + __expf(v - nm);
      mx[r] = nm;
    }
  }

  // reduce (max,sum) across the 16 lanes that share each row
#pragma unroll
  for (int r = 0; r < 8; ++r) {
#pragma unroll
    for (int d = 1; d < 16; d <<= 1) {
      float om = __shfl_xor(mx[r], d, 16);
      float os = __shfl_xor(sm[r], d, 16);
      float nm = fmaxf(mx[r], om);
      sm[r] = sm[r] * __expf(mx[r] - nm) + os * __expf(om - nm);
      mx[r] = nm;
    }
  }
  if (mrow == 0) {
#pragma unroll
    for (int r = 0; r < 8; ++r) {
      pm[ww][half * 8 + r] = mx[r];
      ps[ww][half * 8 + r] = sm[r];
    }
  }
  __syncthreads();
  if (ww == 0 && lane < 16) {
    float m = pm[0][lane], s = ps[0][lane];
#pragma unroll
    for (int k = 1; k < 4; ++k) {
      float om = pm[k][lane], os = ps[k][lane];
      float nm = fmaxf(m, om);
      s = s * __expf(m - nm) + os * __expf(om - nm);
      m = nm;
    }
    rowmax[i0 + lane]   = m;
    rowscale[i0 + lane] = 1.0f / s;
  }
}

// ---------------------------------------------------------------------------
// Kernel 3: x6 = x5 @ softmax(S). Block = 64 output columns (4 j-tiles),
// 16 waves (one 16-row c-block each). Waves 0..7 produce the 32x16 P tile
// (2x WMMA + exp) into LDS in B-operand layout; the block's 256x32 A tile of
// xb is DMA'd into LDS by the Tensor Data Mover (double-buffered, TENSORcnt);
// all waves consume with 4 accumulating WMMAs per 32-deep i-step.
// ---------------------------------------------------------------------------
__global__ __launch_bounds__(512) void k_attn(const __bf16* __restrict__ ybf,
                                              const __bf16* __restrict__ xb,
                                              const float* __restrict__ rowmax,
                                              const float* __restrict__ rowscale,
                                              float* __restrict__ x6) {
  __shared__ u32x4 ldsP[512];                          // [buf2][jt4][n16] 8KB
  __shared__ __align__(128) unsigned char ldsA[2][16384];  // 256 rows x 64B
  const int lane  = threadIdx.x & 31;
  const int ww    = threadIdx.x >> 5;
  const int mrow  = lane & 15;
  const int half  = lane >> 4;
  const int jbase = blockIdx.x * 64;
  const int c0    = ww * 16;

  f32x8 acc[4] = {};

  const int pjt  = ww >> 1;            // producer's j-tile 0..3
  const int psub = ww & 1;             // producer's i sub-block 0..1
  BV Bp;                               // y-column tile: invariant over i loop
  if (ww < 8) {
    const char* bb = (const char*)ybf +
                     (size_t)(jbase + pjt * 16 + mrow) * 64 + half * 16;
    Bp.q[0] = *(const u32x4*)bb;
    Bp.q[1] = *(const u32x4*)(bb + 32);
  }

  const unsigned int ldsA0 = (unsigned int)(uintptr_t)&ldsA[0][0];
  const unsigned int ldsA1 = (unsigned int)(uintptr_t)&ldsA[1][0];
  if (ww == 0) tdm_load_A(xb, 0, ldsA0);          // prime tile for it=0

  for (int it = 0; it < 288; ++it) {
    const int i0  = it * 32;
    const int buf = it & 1;

    if (ww < 8) {
      // S sub-tile: rows i0+psub*16.., cols jbase+pjt*16..
      const char* ab = (const char*)ybf +
                       (size_t)(i0 + psub * 16 + mrow) * 64 + half * 16;
      BV Ap;
      Ap.q[0] = *(const u32x4*)ab;
      Ap.q[1] = *(const u32x4*)(ab + 32);
      f32x8 z = {};
      f32x8 s = __builtin_amdgcn_wmma_f32_16x16x32_bf16(false, Ap.v, false, Bp.v,
                                                        (short)0, z, false, false);
      const float* rmp = rowmax   + i0 + psub * 16 + half * 8;
      const float* rsp = rowscale + i0 + psub * 16 + half * 8;
      f32x4 rm0 = *(const f32x4*)rmp;
      f32x4 rm1 = *(const f32x4*)(rmp + 4);
      f32x4 rs0 = *(const f32x4*)rsp;
      f32x4 rs1 = *(const f32x4*)(rsp + 4);
      float p[8];
#pragma unroll
      for (int r = 0; r < 4; ++r) {
        p[r]     = __expf(s[r]     - rm0[r]) * rs0[r];
        p[r + 4] = __expf(s[r + 4] - rm1[r]) * rs1[r];
      }
      u32x4 pk;
      pk.x = pack_bf16(p[0], p[1]);
      pk.y = pack_bf16(p[2], p[3]);
      pk.z = pack_bf16(p[4], p[5]);
      pk.w = pack_bf16(p[6], p[7]);
      // P[k][n] bf16 at lds bytes: jt*1024 + n*64 + k*2 ; k = psub*16+half*8+r
      ldsP[buf * 256 + pjt * 64 + mrow * 4 + psub * 2 + half] = pk;
    }
    // A-tile for this iteration must have landed before the barrier publishes
    if (ww == 0) __builtin_amdgcn_s_wait_tensorcnt(0);
    __syncthreads();
    // Kick DMA for it+1 into the other buffer; safe: its previous readers
    // (iteration it-1) all finished before the barrier we just passed.
    if (ww == 0 && it + 1 < 288)
      tdm_load_A(xb, i0 + 32, buf ? ldsA0 : ldsA1);

    // consumer: A = xb rows c0..c0+15, k = i0..i0+31 (from TDM-staged LDS)
    BV Ac;
    const unsigned char* la = &ldsA[buf][(unsigned)(c0 + mrow) * 64 + half * 16];
    Ac.q[0] = *(const u32x4*)la;
    Ac.q[1] = *(const u32x4*)(la + 32);
#pragma unroll
    for (int jt = 0; jt < 4; ++jt) {
      BV Bc;
      int idx = buf * 256 + jt * 64 + mrow * 4 + half;
      Bc.q[0] = ldsP[idx];
      Bc.q[1] = ldsP[idx + 2];
      acc[jt] = __builtin_amdgcn_wmma_f32_16x16x32_bf16(false, Ac.v, false, Bc.v,
                                                        (short)0, acc[jt],
                                                        false, false);
    }
  }

#pragma unroll
  for (int jt = 0; jt < 4; ++jt) {
#pragma unroll
    for (int r = 0; r < 8; ++r) {
      int c = c0 + half * 8 + r;       // C/D layout: M = r + 8*half
      int j = jbase + jt * 16 + mrow;  // N = lane%16
      x6[(size_t)c * NN + j] = acc[jt][r];
    }
  }
}

// ---------------------------------------------------------------------------
// Kernel 4: out = softmax over W of (x6 + x). One wave per (c,h) row of 96.
// ---------------------------------------------------------------------------
__global__ __launch_bounds__(256) void k_softmax_w(const float* __restrict__ x6,
                                                   const float* __restrict__ x,
                                                   float* __restrict__ out) {
  const int lane = threadIdx.x & 31;
  const int ww   = threadIdx.x >> 5;
  const int row  = blockIdx.x * 8 + ww;         // c*96 + h
  const size_t base = (size_t)row * WWID;
  float v0 = x6[base + lane     ] + x[base + lane     ];
  float v1 = x6[base + lane + 32] + x[base + lane + 32];
  float v2 = x6[base + lane + 64] + x[base + lane + 64];
  float m = fmaxf(v0, fmaxf(v1, v2));
#pragma unroll
  for (int d = 1; d < 32; d <<= 1) m = fmaxf(m, __shfl_xor(m, d, 32));
  float e0 = __expf(v0 - m), e1 = __expf(v1 - m), e2 = __expf(v2 - m);
  float s = e0 + e1 + e2;
#pragma unroll
  for (int d = 1; d < 32; d <<= 1) s += __shfl_xor(s, d, 32);
  float inv = 1.0f / s;
  out[base + lane     ] = e0 * inv;
  out[base + lane + 32] = e1 * inv;
  out[base + lane + 64] = e2 * inv;
}

// ---------------------------------------------------------------------------
extern "C" void kernel_launch(void* const* d_in, const int* in_sizes, int n_in,
                              void* d_out, int out_size, void* d_ws, size_t ws_size,
                              hipStream_t stream) {
  const float* x = (const float*)d_in[0];
  const float* w = (const float*)d_in[1];
  const float* b = (const float*)d_in[2];
  float* out = (float*)d_out;
  char* ws = (char*)d_ws;

  __bf16* ybf      = (__bf16*)(ws);                 //   576 KB  [9216][32] bf16
  float*  rowmax   = (float*)(ws + 589824);         //    36 KB
  float*  rowscale = (float*)(ws + 626688);         //    36 KB
  __bf16* xb       = (__bf16*)(ws + 663552);        //  4608 KB  [256][9216] bf16
  float*  x6       = (float*)(ws + 5382144);        //  9216 KB  [256][9216] f32

  k_conv     <<< 288, 256, 0, stream>>>(x, w, b, ybf, xb);
  k_rowstats <<< 576, 128, 0, stream>>>(ybf, rowmax, rowscale);
  k_attn     <<< 144, 512, 0, stream>>>(ybf, xb, rowmax, rowscale, x6);
  k_softmax_w<<<3072, 256, 0, stream>>>(x6, x, out);
}
